// VideoSwinTransformer_1803886264400
// MI455X (gfx1250) — matrix operand order repaired and checked by reading, not compiled, not run on hardware
//
#include <hip/hip_runtime.h>
#include <hip/hip_bf16.h>
#include <math.h>

typedef __attribute__((ext_vector_type(16))) _Float16 v16h;
typedef __attribute__((ext_vector_type(8)))  float    v8f;
typedef _Float16 half_t;

#define DEVINL __device__ __forceinline__

// ---------------- problem constants ----------------
#define BB      2
#define TT      16
#define HH      56
#define WWDIM   56
#define C_DIM   96
#define NHEADS  3
#define HDIM    32
#define NTOK    392      // 8*7*7 tokens per window
#define NPAD    400      // padded to 25 * 16
#define NPAD2   416      // padded to 13 * 32 (K dim of P@V)
#define NWIN    256      // B * nW = 2 * 128
#define TOKENS  (BB*TT*HH*WWDIM)   // 100352
#define FFN     384

// ---------------- WMMA fragment helpers (wave32, gfx1250 layouts) ----------------
union FragU { v16h h; unsigned int u[8]; };

// A fragment: 16x32 f16, row-major source, K contiguous.
// lanes 0-15: row=lane, v0..3 = K0..7, v4..7 = K16..23
// lanes 16-31: row=lane-16, v0..3 = K8..15, v4..7 = K24..31
DEVINL v16h ld_fragA(const half_t* __restrict__ p0, int ld) {
  int lane = threadIdx.x & 31;
  int row  = lane & 15;
  int k0   = (lane & 16) ? 8 : 0;
  const half_t* p = p0 + (size_t)row * ld;
  FragU f;
#pragma unroll
  for (int j = 0; j < 4; ++j) {
    f.u[j]     = *(const unsigned int*)(p + k0 + 2*j);
    f.u[j + 4] = *(const unsigned int*)(p + k0 + 16 + 2*j);
  }
  return f.h;
}

// B fragment: 32x16 f16, loaded from B^T (row-major, K contiguous).
// lanes 0-15: col=lane, v0..3 = K0..7, v4..7 = K16..23 ; lanes 16-31: col=lane-16, v0..3 = K8..15, v4..7 = K24..31
DEVINL v16h ld_fragB(const half_t* __restrict__ p0, int ld) {
  int lane = threadIdx.x & 31;
  int col  = lane & 15;
  int k0   = (lane & 16) ? 8 : 0;
  const half_t* p = p0 + (size_t)col * ld + k0;
  FragU f;
#pragma unroll
  for (int j = 0; j < 4; ++j) {
    f.u[j]     = *(const unsigned int*)(p + 2*j);
    f.u[j + 4] = *(const unsigned int*)(p + 16 + 2*j);
  }
  return f.h;
}

DEVINL v8f wmma_f16(v16h a, v16h b, v8f c) {
  return __builtin_amdgcn_wmma_f32_16x16x32_f16(false, a, false, b, (short)0, c, false, false);
}

// 32x32 output tile per wave: 2 A frags x 2 B frags, 4 WMMAs per K step.
DEVINL void gemm_32x32(const half_t* __restrict__ Ab, int lda,
                       const half_t* __restrict__ Bb, int ldb,
                       int K, v8f acc[2][2]) {
  v8f z = {0.f,0.f,0.f,0.f,0.f,0.f,0.f,0.f};
  acc[0][0] = z; acc[0][1] = z; acc[1][0] = z; acc[1][1] = z;
  for (int kk = 0; kk < K; kk += 32) {
    v16h a0 = ld_fragA(Ab + kk, lda);
    v16h a1 = ld_fragA(Ab + (size_t)16 * lda + kk, lda);
    v16h b0 = ld_fragB(Bb + kk, ldb);
    v16h b1 = ld_fragB(Bb + (size_t)16 * ldb + kk, ldb);
    acc[0][0] = wmma_f16(a0, b0, acc[0][0]);
    acc[0][1] = wmma_f16(a0, b1, acc[0][1]);
    acc[1][0] = wmma_f16(a1, b0, acc[1][0]);
    acc[1][1] = wmma_f16(a1, b1, acc[1][1]);
  }
}

// ---------------- kernel 1: weight transpose + f16 convert ----------------
// src is K x N (row-major); dst is N x K (B^T row-major, K contiguous), f16.
__global__ void k_transpose_f16(const float* __restrict__ src, half_t* __restrict__ dst,
                                int K, int N) {
  int i = blockIdx.x * blockDim.x + threadIdx.x;
  if (i >= K * N) return;
  int n = i / K, kk = i % K;
  dst[(size_t)n * K + kk] = (half_t)src[(size_t)kk * N + n];
}

// ---------------- kernel 2: LN1 + roll(-4,-3,-3) + window partition -> f16 ----------------
__global__ void k_ln1_window(const float* __restrict__ x, const float* __restrict__ g,
                             const float* __restrict__ b, half_t* __restrict__ wf16) {
  int wid = blockIdx.x * (blockDim.x >> 5) + (threadIdx.x >> 5);
  if (wid >= NWIN * NPAD) return;
  int lane = threadIdx.x & 31;
  int win = wid / NPAD, n = wid % NPAD;
  half_t* o = wf16 + (size_t)wid * C_DIM;
  if (n >= NTOK) {   // zero padding rows
#pragma unroll
    for (int c = lane; c < C_DIM; c += 32) o[c] = (half_t)0.f;
    return;
  }
  int bidx = win >> 7, r = win & 127;
  int wd = r >> 6, wh = (r >> 3) & 7, ww = r & 7;
  int ld = n / 49, rem = n % 49, lh = rem / 7, lw = rem % 7;
  int d = (wd * 8 + ld + 4) & 15;
  int h = wh * 7 + lh + 3; if (h >= HH) h -= HH;
  int w = ww * 7 + lw + 3; if (w >= WWDIM) w -= WWDIM;
  const float* xr = x + ((((size_t)bidx * TT + d) * HH + h) * WWDIM + w) * C_DIM;
  float v0 = xr[lane], v1 = xr[lane + 32], v2 = xr[lane + 64];
  float s = v0 + v1 + v2;
#pragma unroll
  for (int m = 16; m >= 1; m >>= 1) s += __shfl_xor(s, m, 32);
  float mu = s * (1.f / 96.f);
  float d0 = v0 - mu, d1 = v1 - mu, d2 = v2 - mu;
  float q = d0 * d0 + d1 * d1 + d2 * d2;
#pragma unroll
  for (int m = 16; m >= 1; m >>= 1) q += __shfl_xor(q, m, 32);
  float rs = rsqrtf(q * (1.f / 96.f) + 1e-5f);
  o[lane]      = (half_t)(d0 * rs * g[lane]      + b[lane]);
  o[lane + 32] = (half_t)(d1 * rs * g[lane + 32] + b[lane + 32]);
  o[lane + 64] = (half_t)(d2 * rs * g[lane + 64] + b[lane + 64]);
}

// ---------------- kernel 3: rel-pos bias + shift mask table ----------------
// tab[cls][head][n][m] (m-stride NPAD), cls = (wd!=0)<<2 | (wh==7)<<1 | (ww==7)
__global__ void k_biasmask(const float* __restrict__ rpb, float* __restrict__ tab) {
  size_t i = (size_t)blockIdx.x * blockDim.x + threadIdx.x;
  const size_t total = (size_t)8 * NHEADS * NTOK * NTOK;
  if (i >= total) return;
  int m = (int)(i % NTOK); size_t t = i / NTOK;
  int n = (int)(t % NTOK); t /= NTOK;
  int h = (int)(t % NHEADS); int cls = (int)(t / NHEADS);
  int d1 = n / 49, h1 = (n % 49) / 7, w1 = n % 7;
  int d2 = m / 49, h2 = (m % 49) / 7, w2 = m % 7;
  int idx = (d1 - d2 + 7) * 169 + (h1 - h2 + 6) * 13 + (w1 - w2 + 6);
  float v = rpb[(size_t)idx * NHEADS + h];
  int ad = (cls >> 2) & 1, ah = (cls >> 1) & 1, aw = cls & 1;
  int rd1 = ad ? ((d1 < 4) ? 1 : 2) : 0, rd2 = ad ? ((d2 < 4) ? 1 : 2) : 0;
  int rh1 = ah ? ((h1 < 4) ? 1 : 2) : 0, rh2 = ah ? ((h2 < 4) ? 1 : 2) : 0;
  int rw1 = aw ? ((w1 < 4) ? 1 : 2) : 0, rw2 = aw ? ((w2 < 4) ? 1 : 2) : 0;
  if (rd1 != rd2 || rh1 != rh2 || rw1 != rw2) v -= 100.f;
  tab[((size_t)(cls * NHEADS + h) * NTOK + n) * NPAD + m] = v;
}

// ---------------- kernel 4: QKV GEMM + scatter to q / k / v^T ----------------
__global__ __launch_bounds__(128)
void k_qkv(const half_t* __restrict__ A, const half_t* __restrict__ WT,
           const float* __restrict__ bias,
           half_t* __restrict__ q, half_t* __restrict__ kb, half_t* __restrict__ vT) {
  const int MT = (NWIN * NPAD) / 32, NT = (3 * C_DIM) / 32;   // 3200 x 9
  int wid = blockIdx.x * (blockDim.x >> 5) + (threadIdx.x >> 5);
  if (wid >= MT * NT) return;
  int mt = wid / NT, nt = wid % NT;
  v8f acc[2][2];
  gemm_32x32(A + (size_t)mt * 32 * C_DIM, C_DIM,
             WT + (size_t)nt * 32 * C_DIM, C_DIM, C_DIM, acc);
  int lane = threadIdx.x & 31, col = lane & 15, rb = (lane >> 4) * 8;
#pragma unroll
  for (int mi = 0; mi < 2; ++mi)
#pragma unroll
    for (int ni = 0; ni < 2; ++ni)
#pragma unroll
      for (int j = 0; j < 8; ++j) {
        int row = mt * 32 + mi * 16 + rb + j;
        int c   = nt * 32 + ni * 16 + col;
        float v = acc[mi][ni][j] + bias[c];
        int win = row / NPAD, n = row % NPAD;
        if (c < C_DIM) {
          int hd = c >> 5, dd = c & 31;
          q[(((size_t)win * NHEADS + hd) * NPAD + n) * HDIM + dd] =
              (half_t)(v * 0.17677669529663687f);           // * 32^-0.5
        } else if (c < 2 * C_DIM) {
          int cc = c - C_DIM; int hd = cc >> 5, dd = cc & 31;
          kb[(((size_t)win * NHEADS + hd) * NPAD + n) * HDIM + dd] = (half_t)v;
        } else {
          int cc = c - 2 * C_DIM; int hd = cc >> 5, dd = cc & 31;
          vT[(((size_t)win * NHEADS + hd) * HDIM + dd) * NPAD2 + n] = (half_t)v;
        }
      }
}

// ---------------- kernel 5: windowed attention (one WG per win/head/16-row block) ----------------
__global__ __launch_bounds__(128)
void k_attn(const half_t* __restrict__ q, const half_t* __restrict__ kmat,
            const half_t* __restrict__ vT, const float* __restrict__ tab,
            half_t* __restrict__ o) {
  __shared__ float  sS[16 * NPAD];      // fp32 scores
  __shared__ half_t sP[16 * NPAD2];     // unnormalized exp (f16)
  __shared__ float  pacc[4 * 256];      // stage-2 partials
  __shared__ float  red[16 * 8];
  __shared__ float  rowmax[16], rowsum[16];

  int blk = blockIdx.x;
  int qt = blk % 25; int t2 = blk / 25;
  int head = t2 % NHEADS; int win = t2 / NHEADS;
  int n0 = qt * 16;
  int tid = threadIdx.x, wv = tid >> 5, lane = tid & 31;
  int col = lane & 15, rb = (lane >> 4) * 8;

  int r = win & 127;
  int wd = r >> 6, wh = (r >> 3) & 7, ww = r & 7;
  int cls = ((wd != 0) ? 4 : 0) | ((wh == 7) ? 2 : 0) | ((ww == 7) ? 1 : 0);
  const float* brow = tab + (size_t)(cls * NHEADS + head) * NTOK * NPAD;

  const half_t* qh = q    + ((size_t)win * NHEADS + head) * NPAD * HDIM;
  const half_t* kh = kmat + ((size_t)win * NHEADS + head) * NPAD * HDIM;
  const half_t* vh = vT   + ((size_t)win * NHEADS + head) * HDIM * NPAD2;

  // ---- stage 1: S = q_blk @ K^T (+bias +mask), K=hd=32 (one WMMA per 16x16 tile)
  v16h afr = ld_fragA(qh + (size_t)n0 * HDIM, HDIM);
  for (int ct = wv; ct < 25; ct += 4) {
    v16h bfr = ld_fragB(kh + (size_t)ct * 16 * HDIM, HDIM);
    v8f z = {0.f,0.f,0.f,0.f,0.f,0.f,0.f,0.f};
    v8f s = wmma_f16(afr, bfr, z);
    int m = ct * 16 + col;
#pragma unroll
    for (int j = 0; j < 8; ++j) {
      int n = n0 + rb + j;
      float sc = (m >= NTOK || n >= NTOK) ? -1e30f
                                          : s[j] + brow[(size_t)n * NPAD + m];
      sS[(rb + j) * NPAD + m] = sc;
    }
  }
  __syncthreads();

  // ---- softmax over 400 cols (8 threads per row)
  { int row = tid >> 3, sub = tid & 7;
    float lm = -1e30f;
    for (int m = sub; m < NPAD; m += 8) lm = fmaxf(lm, sS[row * NPAD + m]);
    red[row * 8 + sub] = lm; }
  __syncthreads();
  if ((tid & 7) == 0) { int row = tid >> 3; float mx = red[row * 8];
#pragma unroll
    for (int j = 1; j < 8; ++j) mx = fmaxf(mx, red[row * 8 + j]);
    rowmax[row] = mx; }
  __syncthreads();
  { int row = tid >> 3, sub = tid & 7; float mx = rowmax[row]; float ls = 0.f;
    for (int m = sub; m < NPAD; m += 8) {
      float e = __expf(sS[row * NPAD + m] - mx);
      sP[row * NPAD2 + m] = (half_t)e;
      ls += e;
    }
    red[row * 8 + sub] = ls; }
  for (int i = tid; i < 16 * 16; i += 128) {    // zero K-pad cols 400..415
    int rr = i >> 4; sP[rr * NPAD2 + NPAD + (i & 15)] = (half_t)0.f;
  }
  __syncthreads();
  if ((tid & 7) == 0) { int row = tid >> 3; float s = 0.f;
#pragma unroll
    for (int j = 0; j < 8; ++j) s += red[row * 8 + j];
    rowsum[row] = s; }
  __syncthreads();

  // ---- stage 2: O = P @ V, K=416 -> 13 WMMA steps split across wave pairs
  { int tile = wv >> 1;                 // output cols tile*16 .. +15 (of 32)
    int hf = wv & 1;
    int ks0 = hf ? 7 : 0, ks1 = hf ? 13 : 7;
    v8f accv = {0.f,0.f,0.f,0.f,0.f,0.f,0.f,0.f};
    for (int ks = ks0; ks < ks1; ++ks) {
      v16h pa = ld_fragA((const half_t*)sP + ks * 32, NPAD2);
      v16h vb = ld_fragB(vh + (size_t)(tile * 16) * NPAD2 + ks * 32, NPAD2);
      accv = wmma_f16(pa, vb, accv);
    }
#pragma unroll
    for (int j = 0; j < 8; ++j) pacc[wv * 256 + (rb + j) * 16 + col] = accv[j]; }
  __syncthreads();
  for (int i = tid; i < 512; i += 128) {
    int tile = i >> 8, idx = i & 255, rr = idx >> 4, cc = idx & 15;
    float v = (pacc[(tile * 2) * 256 + idx] + pacc[(tile * 2 + 1) * 256 + idx]) / rowsum[rr];
    int n = n0 + rr;
    o[((size_t)win * NPAD + n) * C_DIM + head * HDIM + tile * 16 + cc] = (half_t)v;
  }
}

// ---------------- kernel 6: proj GEMM + window reverse + roll(+4,+3,+3) + residual ----------------
__global__ __launch_bounds__(128)
void k_proj(const half_t* __restrict__ A, const half_t* __restrict__ WT,
            const float* __restrict__ bias, const float* __restrict__ x,
            float* __restrict__ x1) {
  const int MT = (NWIN * NPAD) / 32, NT = C_DIM / 32;   // 3200 x 3
  int wid = blockIdx.x * (blockDim.x >> 5) + (threadIdx.x >> 5);
  if (wid >= MT * NT) return;
  int mt = wid / NT, nt = wid % NT;
  v8f acc[2][2];
  gemm_32x32(A + (size_t)mt * 32 * C_DIM, C_DIM,
             WT + (size_t)nt * 32 * C_DIM, C_DIM, C_DIM, acc);
  int lane = threadIdx.x & 31, col = lane & 15, rb = (lane >> 4) * 8;
#pragma unroll
  for (int mi = 0; mi < 2; ++mi)
#pragma unroll
    for (int ni = 0; ni < 2; ++ni)
#pragma unroll
      for (int j = 0; j < 8; ++j) {
        int row = mt * 32 + mi * 16 + rb + j;
        int c   = nt * 32 + ni * 16 + col;
        int win = row / NPAD, n = row % NPAD;
        if (n >= NTOK) continue;
        int bidx = win >> 7, rr = win & 127;
        int wd = rr >> 6, wh = (rr >> 3) & 7, ww = rr & 7;
        int ld = n / 49, rem = n % 49, lh = rem / 7, lw = rem % 7;
        int d = (wd * 8 + ld + 4) & 15;
        int h = wh * 7 + lh + 3; if (h >= HH) h -= HH;
        int w = ww * 7 + lw + 3; if (w >= WWDIM) w -= WWDIM;
        size_t gi = ((((size_t)bidx * TT + d) * HH + h) * WWDIM + w) * C_DIM + c;
        x1[gi] = x[gi] + acc[mi][ni][j] + bias[c];
      }
}

// ---------------- kernel 7: LN2 ----------------
__global__ void k_ln2(const float* __restrict__ x1, const float* __restrict__ g,
                      const float* __restrict__ b, half_t* __restrict__ o) {
  int wid = blockIdx.x * (blockDim.x >> 5) + (threadIdx.x >> 5);
  if (wid >= TOKENS) return;
  int lane = threadIdx.x & 31;
  const float* xr = x1 + (size_t)wid * C_DIM;
  float v0 = xr[lane], v1 = xr[lane + 32], v2 = xr[lane + 64];
  float s = v0 + v1 + v2;
#pragma unroll
  for (int m = 16; m >= 1; m >>= 1) s += __shfl_xor(s, m, 32);
  float mu = s * (1.f / 96.f);
  float d0 = v0 - mu, d1 = v1 - mu, d2 = v2 - mu;
  float q = d0 * d0 + d1 * d1 + d2 * d2;
#pragma unroll
  for (int m = 16; m >= 1; m >>= 1) q += __shfl_xor(q, m, 32);
  float rs = rsqrtf(q * (1.f / 96.f) + 1e-5f);
  half_t* orow = o + (size_t)wid * C_DIM;
  orow[lane]      = (half_t)(d0 * rs * g[lane]      + b[lane]);
  orow[lane + 32] = (half_t)(d1 * rs * g[lane + 32] + b[lane + 32]);
  orow[lane + 64] = (half_t)(d2 * rs * g[lane + 64] + b[lane + 64]);
}

// ---------------- kernel 8: fc1 GEMM + bias + exact GELU -> f16 ----------------
__global__ __launch_bounds__(128)
void k_fc1(const half_t* __restrict__ A, const half_t* __restrict__ WT,
           const float* __restrict__ bias, half_t* __restrict__ h1) {
  const int MT = TOKENS / 32, NT = FFN / 32;   // 3136 x 12
  int wid = blockIdx.x * (blockDim.x >> 5) + (threadIdx.x >> 5);
  if (wid >= MT * NT) return;
  int mt = wid / NT, nt = wid % NT;
  v8f acc[2][2];
  gemm_32x32(A + (size_t)mt * 32 * C_DIM, C_DIM,
             WT + (size_t)nt * 32 * C_DIM, C_DIM, C_DIM, acc);
  int lane = threadIdx.x & 31, col = lane & 15, rb = (lane >> 4) * 8;
#pragma unroll
  for (int mi = 0; mi < 2; ++mi)
#pragma unroll
    for (int ni = 0; ni < 2; ++ni)
#pragma unroll
      for (int j = 0; j < 8; ++j) {
        int row = mt * 32 + mi * 16 + rb + j;
        int c   = nt * 32 + ni * 16 + col;
        float v = acc[mi][ni][j] + bias[c];
        v = 0.5f * v * (1.f + erff(v * 0.70710678118654752f));
        h1[(size_t)row * FFN + c] = (half_t)v;
      }
}

// ---------------- kernel 9: fc2 GEMM + bias + residual -> out ----------------
__global__ __launch_bounds__(128)
void k_fc2(const half_t* __restrict__ A, const half_t* __restrict__ WT,
           const float* __restrict__ bias, const float* __restrict__ x1,
           float* __restrict__ out) {
  const int MT = TOKENS / 32, NT = C_DIM / 32;   // 3136 x 3
  int wid = blockIdx.x * (blockDim.x >> 5) + (threadIdx.x >> 5);
  if (wid >= MT * NT) return;
  int mt = wid / NT, nt = wid % NT;
  v8f acc[2][2];
  gemm_32x32(A + (size_t)mt * 32 * FFN, FFN,
             WT + (size_t)nt * 32 * FFN, FFN, FFN, acc);
  int lane = threadIdx.x & 31, col = lane & 15, rb = (lane >> 4) * 8;
#pragma unroll
  for (int mi = 0; mi < 2; ++mi)
#pragma unroll
    for (int ni = 0; ni < 2; ++ni)
#pragma unroll
      for (int j = 0; j < 8; ++j) {
        int row = mt * 32 + mi * 16 + rb + j;
        int c   = nt * 32 + ni * 16 + col;
        size_t gi = (size_t)row * C_DIM + c;
        out[gi] = x1[gi] + acc[mi][ni][j] + bias[c];
      }
}

// ---------------- host launcher ----------------
extern "C" void kernel_launch(void* const* d_in, const int* in_sizes, int n_in,
                              void* d_out, int out_size, void* d_ws, size_t ws_size,
                              hipStream_t stream) {
  (void)in_sizes; (void)n_in; (void)out_size; (void)ws_size;
  const float* x      = (const float*)d_in[0];
  const float* ln1_g  = (const float*)d_in[1];
  const float* ln1_b  = (const float*)d_in[2];
  const float* qkv_w  = (const float*)d_in[3];
  const float* qkv_b  = (const float*)d_in[4];
  const float* rpb    = (const float*)d_in[5];
  const float* proj_w = (const float*)d_in[6];
  const float* proj_b = (const float*)d_in[7];
  const float* ln2_g  = (const float*)d_in[8];
  const float* ln2_b  = (const float*)d_in[9];
  const float* fc1_w  = (const float*)d_in[10];
  const float* fc1_b  = (const float*)d_in[11];
  const float* fc2_w  = (const float*)d_in[12];
  const float* fc2_b  = (const float*)d_in[13];
  float* out = (float*)d_out;

  char* ws = (char*)d_ws;
  size_t off = 0;
  auto alloc = [&](size_t bytes) {
    size_t p = off;
    off += (bytes + 255) & ~(size_t)255;
    return p;
  };
  half_t* wf16  = (half_t*)(ws + alloc((size_t)NWIN * NPAD * C_DIM * 2));
  half_t* qbuf  = (half_t*)(ws + alloc((size_t)NWIN * NHEADS * NPAD * HDIM * 2));
  half_t* kbuf  = (half_t*)(ws + alloc((size_t)NWIN * NHEADS * NPAD * HDIM * 2));
  size_t vt_bytes = (size_t)NWIN * NHEADS * HDIM * NPAD2 * 2;
  half_t* vtbuf = (half_t*)(ws + alloc(vt_bytes));
  half_t* obuf  = (half_t*)(ws + alloc((size_t)NWIN * NPAD * C_DIM * 2));
  float*  x1    = (float*) (ws + alloc((size_t)TOKENS * C_DIM * 4));
  half_t* ln2h  = (half_t*)(ws + alloc((size_t)TOKENS * C_DIM * 2));
  half_t* h1    = (half_t*)(ws + alloc((size_t)TOKENS * FFN * 2));
  float*  tab   = (float*) (ws + alloc((size_t)8 * NHEADS * NTOK * NPAD * 4));
  half_t* qkvT  = (half_t*)(ws + alloc((size_t)3 * C_DIM * C_DIM * 2));
  half_t* projT = (half_t*)(ws + alloc((size_t)C_DIM * C_DIM * 2));
  half_t* fc1T  = (half_t*)(ws + alloc((size_t)FFN * C_DIM * 2));
  half_t* fc2T  = (half_t*)(ws + alloc((size_t)C_DIM * FFN * 2));

  // 1) weights -> f16 B^T
  k_transpose_f16<<<(C_DIM * 3 * C_DIM + 255) / 256, 256, 0, stream>>>(qkv_w, qkvT, C_DIM, 3 * C_DIM);
  k_transpose_f16<<<(C_DIM * C_DIM + 255) / 256, 256, 0, stream>>>(proj_w, projT, C_DIM, C_DIM);
  k_transpose_f16<<<(C_DIM * FFN + 255) / 256, 256, 0, stream>>>(fc1_w, fc1T, C_DIM, FFN);
  k_transpose_f16<<<(FFN * C_DIM + 255) / 256, 256, 0, stream>>>(fc2_w, fc2T, FFN, C_DIM);

  // 2) bias+mask table
  k_biasmask<<<(unsigned)(((size_t)8 * NHEADS * NTOK * NTOK + 255) / 256), 256, 0, stream>>>(rpb, tab);

  // 3) LN1 + roll + window partition
  k_ln1_window<<<(NWIN * NPAD + 3) / 4, 128, 0, stream>>>(x, ln1_g, ln1_b, wf16);

  // 4) zero v^T padding, QKV GEMM
  hipMemsetAsync(vtbuf, 0, vt_bytes, stream);
  { int waves = ((NWIN * NPAD) / 32) * ((3 * C_DIM) / 32);
    k_qkv<<<(waves + 3) / 4, 128, 0, stream>>>(wf16, qkvT, qkv_b, qbuf, kbuf, vtbuf); }

  // 5) attention
  k_attn<<<NWIN * NHEADS * 25, 128, 0, stream>>>(qbuf, kbuf, vtbuf, tab, obuf);

  // 6) proj + window reverse + roll + residual
  { int waves = ((NWIN * NPAD) / 32) * (C_DIM / 32);
    k_proj<<<(waves + 3) / 4, 128, 0, stream>>>(obuf, projT, proj_b, x, x1); }

  // 7) LN2
  k_ln2<<<(TOKENS + 3) / 4, 128, 0, stream>>>(x1, ln2_g, ln2_b, ln2h);

  // 8) fc1 + GELU
  { int waves = (TOKENS / 32) * (FFN / 32);
    k_fc1<<<(waves + 3) / 4, 128, 0, stream>>>(ln2h, fc1T, fc1_b, h1); }

  // 9) fc2 + residual -> out
  { int waves = (TOKENS / 32) * (C_DIM / 32);
    k_fc2<<<(waves + 3) / 4, 128, 0, stream>>>(h1, fc2T, fc2_b, x1, out); }
}
